// AutoCorrelationLayer_80255758893094
// MI455X (gfx1250) — compile-verified, matrix operand order checked by reading, not run on hardware
//
#include <hip/hip_runtime.h>

#define LSEQ 4096
#define DM   512
#define NH   8
#define DHD  64
#define BB   4
#define KTOP 8

typedef __attribute__((ext_vector_type(16))) __bf16 v16bf;
typedef __attribute__((ext_vector_type(8)))  float  v8f;

union Frag16 { v16bf v; unsigned int u[8]; };

__device__ __forceinline__ unsigned short f2bf_bits(float f) {
  unsigned int u = __builtin_bit_cast(unsigned int, f);
  unsigned int r = u + 0x7FFFu + ((u >> 16) & 1u);   // round-to-nearest-even
  return (unsigned short)(r >> 16);
}
__device__ __forceinline__ float bf2f(unsigned short s) {
  unsigned int u = ((unsigned int)s) << 16;
  return __builtin_bit_cast(float, u);
}

__device__ __forceinline__ v8f wmma_bf16(v16bf a, v16bf b, v8f c) {
  return __builtin_amdgcn_wmma_f32_16x16x32_bf16(false, a, false, b,
                                                 (short)0, c, false, false);
}

// ---------------------------------------------------------------------------
// 16-bit WMMA operand fragment loader (wave32).
// A-layout (16x32, MxK): lane l -> row m = l&15; lanes 0-15 hold K = 0..7 and
// 16..23 (pairs per VGPR), lanes 16-31 hold K = 8..15 and 24..31.
// B fragments use the same per-lane K pattern with lane -> column, so loading
// from an N-major (transposed) matrix uses this same routine. Works for both
// global and LDS-resident sources (address space inferred after inlining).
// ---------------------------------------------------------------------------
__device__ __forceinline__ v16bf load_frag_row(const unsigned short* __restrict__ X,
                                               int ldx, int row0, int k0, int lane) {
  int m  = lane & 15;
  int kb = (lane & 16) ? 8 : 0;
  const unsigned short* p = X + (size_t)(row0 + m) * ldx + k0;
  Frag16 f;
#pragma unroll
  for (int j = 0; j < 4; ++j) {
    f.u[j]     = *(const unsigned int*)(p + kb + 2 * j);        // K = kb+2j, kb+2j+1
    f.u[4 + j] = *(const unsigned int*)(p + 16 + kb + 2 * j);   // K = 16+kb+2j, ...
  }
  return f.v;
}

// ---------------------------------------------------------------------------
// Async DMA: stage 16 KB (128 rows x 64 bf16) of K into LDS.
// 256 threads x 4 x global_load_async_to_lds_b128 (ASYNCcnt-tracked).
// ---------------------------------------------------------------------------
__device__ __forceinline__ void stage_chunk_async(const unsigned short* __restrict__ gsrc,
                                                  unsigned short* lds_dst, int tid) {
  unsigned int base = (unsigned int)(size_t)lds_dst + (unsigned int)tid * 64u;
  const char* g = (const char*)gsrc + (size_t)tid * 64;
#pragma unroll
  for (int j = 0; j < 4; ++j) {
    asm volatile("global_load_async_to_lds_b128 %0, %1, off"
                 :: "v"(base + (unsigned int)(j * 16)), "v"((const void*)(g + j * 16))
                 : "memory");
  }
}
__device__ __forceinline__ void wait_async0() {
  asm volatile("s_wait_asynccnt 0x0" ::: "memory");
}

// ---------------------------------------------------------------------------
// Elementwise f32 -> bf16
// ---------------------------------------------------------------------------
__global__ void cvt_f32_bf16(const float* __restrict__ in, unsigned short* __restrict__ out, int n) {
  int i = blockIdx.x * 256 + threadIdx.x;
  if (i < n) out[i] = f2bf_bits(in[i]);
}

// Convert 512x512 weight f32 (K-major) -> bf16 transposed (N-major): Wt[n*512+k] = W[k*512+n]
__global__ void cvt_w_transpose(const float* __restrict__ in, unsigned short* __restrict__ out) {
  int i = blockIdx.x * 256 + threadIdx.x;     // i = k*512 + n
  int k = i >> 9, n = i & 511;
  out[n * DM + k] = f2bf_bits(in[i]);
}

__global__ void zero_f32(float* __restrict__ p, int n) {
  int i = blockIdx.x * 256 + threadIdx.x;
  if (i < n) p[i] = 0.f;
}

// ---------------------------------------------------------------------------
// Projection GEMM: Y = A(bf16, Mx512) * W(bf16, 512x512, given N-major) + bias.
// One 16x16 output tile per wave, K swept in 16 steps of 32 (bf16 WMMA).
// headMajor: scatter Y[b*L+t, h*64+d] -> out[(b*8+h)*L*64 + t*64 + d] (bf16).
// ---------------------------------------------------------------------------
__global__ void proj_gemm(const unsigned short* __restrict__ A,
                          const unsigned short* __restrict__ Bt,
                          const float* __restrict__ bias,
                          unsigned short* __restrict__ outBf,
                          float* __restrict__ outF,
                          int M, int headMajor) {
  int lane = threadIdx.x & 31;
  int wave = threadIdx.x >> 5;
  int tile = blockIdx.x * 8 + wave;
  const int tiles_n = DM / 16;                 // 32
  int tm = tile / tiles_n, tn = tile % tiles_n;
  int row0 = tm * 16, col0 = tn * 16;
  if (row0 >= M) return;                       // wave-uniform

  v8f acc = {};
#pragma unroll 1
  for (int k0 = 0; k0 < DM; k0 += 32) {
    if (k0 + 32 < DM)
      __builtin_prefetch(A + (size_t)(row0 + (lane & 15)) * DM + k0 + 32, 0, 0);
    v16bf a = load_frag_row(A,  DM, row0, k0, lane);
    v16bf b = load_frag_row(Bt, DM, col0, k0, lane);
    acc = wmma_bf16(a, b, acc);
  }

  int n     = lane & 15;
  int col   = col0 + n;
  int half8 = (lane & 16) ? 8 : 0;
  float bv  = bias[col];
#pragma unroll
  for (int v = 0; v < 8; ++v) {
    int row   = row0 + v + half8;
    float val = acc[v] + bv;
    if (headMajor) {
      int b_ = row >> 12, t = row & (LSEQ - 1);
      int h  = col >> 6,  d = col & 63;
      size_t idx = ((size_t)((b_ * NH + h) * LSEQ + t)) * DHD + d;
      outBf[idx] = f2bf_bits(val);
    } else if (outBf) {
      outBf[(size_t)row * DM + col] = f2bf_bits(val);
    } else {
      outF[(size_t)row * DM + col] = val;
    }
  }
}

// ---------------------------------------------------------------------------
// Correlation: corr[bh, tau] += sum_t <q[t,:], k[(t-tau) mod L,:]>
// via S = Q K^T in 16x16 WMMA tiles; each S element scattered to tau = (t-s)&4095
// in an LDS histogram, flushed with global float atomics.
// Block = 8 waves; wave w owns t-rows [strip*128 + w*16, +16) (A-fragments
// loaded once, resident for the whole sweep). K rows are staged block-wide in
// LDS by the async DMA path, double-buffered: while chunk i is multiplied,
// chunk i+1 streams in via global_load_async_to_lds_b128.
// ---------------------------------------------------------------------------
__global__ void corr_kernel(const unsigned short* __restrict__ Qh,
                            const unsigned short* __restrict__ Kh,
                            float* __restrict__ corr) {
  __shared__ float lc[LSEQ];                       // 16 KB tau-histogram
  __shared__ unsigned short kst[2][128 * DHD];     // 2 x 16 KB K staging
  int strip = blockIdx.x & 31;                     // 32 strips of 128 t-rows
  int bh    = blockIdx.x >> 5;                     // 0..31
  const unsigned short* q    = Qh + (size_t)bh * LSEQ * DHD;
  const unsigned short* kmat = Kh + (size_t)bh * LSEQ * DHD;

  for (int i = threadIdx.x; i < LSEQ; i += 256) lc[i] = 0.f;

  int lane  = threadIdx.x & 31;
  int wave  = threadIdx.x >> 5;
  int t0    = strip * 128 + wave * 16;
  int n     = lane & 15;
  int half8 = (lane & 16) ? 8 : 0;

  v16bf a0 = load_frag_row(q, DHD, t0, 0,  lane);  // d = 0..31, resident
  v16bf a1 = load_frag_row(q, DHD, t0, 32, lane);  // d = 32..63, resident

  stage_chunk_async(kmat, &kst[0][0], threadIdx.x);  // prime chunk 0

  int buf = 0;
#pragma unroll 1
  for (int chunk = 0; chunk < LSEQ; chunk += 128) {
    wait_async0();        // my wave's loads for this chunk have landed
    __syncthreads();      // all waves landed => chunk visible, other buf free
    if (chunk + 128 < LSEQ)
      stage_chunk_async(kmat + (size_t)(chunk + 128) * DHD,
                        &kst[buf ^ 1][0], threadIdx.x);
    const unsigned short* kb = &kst[buf][0];
#pragma unroll 1
    for (int st = 0; st < 8; ++st) {
      int sl = st * 16;
      v16bf b0 = load_frag_row(kb, DHD, sl, 0,  lane);   // ds_load path
      v16bf b1 = load_frag_row(kb, DHD, sl, 32, lane);
      v8f acc = {};
      acc = wmma_bf16(a0, b0, acc);
      acc = wmma_bf16(a1, b1, acc);
      int s = chunk + sl + n;
#pragma unroll
      for (int v = 0; v < 8; ++v) {
        int t   = t0 + v + half8;
        int tau = (t - s) & (LSEQ - 1);
        atomicAdd(&lc[tau], acc[v]);                     // ds_add_f32
      }
    }
    buf ^= 1;
  }
  __syncthreads();
  for (int i = threadIdx.x; i < LSEQ; i += 256)
    atomicAdd(&corr[(size_t)bh * LSEQ + i], lc[i]);
}

// ---------------------------------------------------------------------------
// Per-(b,h): top-8 of corr/64 over tau, softmax over the 8 values.
// ---------------------------------------------------------------------------
__global__ void topk_softmax(const float* __restrict__ corr,
                             float* __restrict__ attn,
                             int* __restrict__ delays) {
  __shared__ float vals[LSEQ];
  __shared__ float rv[256];
  __shared__ int   ri[256];
  __shared__ float wsel[KTOP];
  __shared__ int   dsel[KTOP];
  int bh = blockIdx.x;
  const float scale = 1.0f / (float)DHD;       // mean over Dh (matters for softmax)
  for (int i = threadIdx.x; i < LSEQ; i += 256)
    vals[i] = corr[(size_t)bh * LSEQ + i] * scale;
  __syncthreads();

  for (int k = 0; k < KTOP; ++k) {
    float best = -3.4e38f; int bi = 0;
    for (int i = threadIdx.x; i < LSEQ; i += 256)
      if (vals[i] > best) { best = vals[i]; bi = i; }
    rv[threadIdx.x] = best; ri[threadIdx.x] = bi;
    __syncthreads();
    for (int off = 128; off > 0; off >>= 1) {
      if (threadIdx.x < off) {
        if (rv[threadIdx.x + off] > rv[threadIdx.x]) {
          rv[threadIdx.x] = rv[threadIdx.x + off];
          ri[threadIdx.x] = ri[threadIdx.x + off];
        }
      }
      __syncthreads();
    }
    if (threadIdx.x == 0) { wsel[k] = rv[0]; dsel[k] = ri[0]; vals[ri[0]] = -3.4e38f; }
    __syncthreads();
  }

  if (threadIdx.x == 0) {
    float mx = wsel[0];
    for (int k = 1; k < KTOP; ++k) mx = fmaxf(mx, wsel[k]);
    float e[KTOP]; float sum = 0.f;
    for (int k = 0; k < KTOP; ++k) { e[k] = expf(wsel[k] - mx); sum += e[k]; }
    for (int k = 0; k < KTOP; ++k) {
      attn[bh * KTOP + k]   = e[k] / sum;
      delays[bh * KTOP + k] = dsel[k];
    }
  }
}

// ---------------------------------------------------------------------------
// ctx[b, t, h*64+d] = sum_k attn[bh,k] * V[bh, (t - delay_k) & 4095, d]
// ---------------------------------------------------------------------------
__global__ void gather_ctx(const unsigned short* __restrict__ Vh,
                           const float* __restrict__ attn,
                           const int* __restrict__ delays,
                           unsigned short* __restrict__ ctx) {
  int idx = blockIdx.x * 256 + threadIdx.x;    // b*L*512 + t*512 + c
  int c = idx & (DM - 1);
  int t = (idx >> 9) & (LSEQ - 1);
  int b = idx >> 21;                           // 512*4096 = 2^21
  int h = c >> 6, d = c & 63;
  int bh = b * NH + h;
  const unsigned short* vbase = Vh + (size_t)bh * LSEQ * DHD;
  float sum = 0.f;
#pragma unroll
  for (int k = 0; k < KTOP; ++k) {
    float w = attn[bh * KTOP + k];
    int  dl = delays[bh * KTOP + k];
    int src = (t - dl) & (LSEQ - 1);
    sum += w * bf2f(vbase[(size_t)src * DHD + d]);
  }
  ctx[idx] = f2bf_bits(sum);
}

// ---------------------------------------------------------------------------
extern "C" void kernel_launch(void* const* d_in, const int* in_sizes, int n_in,
                              void* d_out, int out_size, void* d_ws, size_t ws_size,
                              hipStream_t stream) {
  const float* x_q  = (const float*)d_in[0];
  const float* x_kv = (const float*)d_in[1];
  const float* Wq   = (const float*)d_in[2];
  const float* bq   = (const float*)d_in[3];
  const float* Wk   = (const float*)d_in[4];
  const float* bk   = (const float*)d_in[5];
  const float* Wv   = (const float*)d_in[6];
  const float* bv   = (const float*)d_in[7];
  const float* Wo   = (const float*)d_in[8];
  const float* bo   = (const float*)d_in[9];
  float* out = (float*)d_out;

  const size_t ML = (size_t)BB * LSEQ;         // 16384
  char* ws = (char*)d_ws;
  size_t off = 0;
  auto alloc = [&](size_t bytes) -> char* {
    char* p = ws + off;
    off += (bytes + 255) & ~(size_t)255;
    return p;
  };
  unsigned short* xq_bf  = (unsigned short*)alloc(ML * DM * 2);
  unsigned short* xkv_bf = (unsigned short*)alloc(ML * DM * 2);
  unsigned short* Wq_t   = (unsigned short*)alloc((size_t)DM * DM * 2);
  unsigned short* Wk_t   = (unsigned short*)alloc((size_t)DM * DM * 2);
  unsigned short* Wv_t   = (unsigned short*)alloc((size_t)DM * DM * 2);
  unsigned short* Wo_t   = (unsigned short*)alloc((size_t)DM * DM * 2);
  unsigned short* Qh     = (unsigned short*)alloc(ML * DM * 2);
  unsigned short* Kh     = (unsigned short*)alloc(ML * DM * 2);
  unsigned short* Vh     = (unsigned short*)alloc(ML * DM * 2);
  unsigned short* ctx    = (unsigned short*)alloc(ML * DM * 2);
  float*          corr   = (float*)alloc((size_t)BB * NH * LSEQ * 4);
  float*          attn   = (float*)alloc((size_t)BB * NH * KTOP * 4);
  int*            delays = (int*)alloc((size_t)BB * NH * KTOP * 4);
  (void)ws_size; (void)in_sizes; (void)n_in; (void)out_size;

  const int nX = (int)(ML * DM);               // 8,388,608
  cvt_f32_bf16<<<nX / 256, 256, 0, stream>>>(x_q,  xq_bf,  nX);
  cvt_f32_bf16<<<nX / 256, 256, 0, stream>>>(x_kv, xkv_bf, nX);
  cvt_w_transpose<<<(DM * DM) / 256, 256, 0, stream>>>(Wq, Wq_t);
  cvt_w_transpose<<<(DM * DM) / 256, 256, 0, stream>>>(Wk, Wk_t);
  cvt_w_transpose<<<(DM * DM) / 256, 256, 0, stream>>>(Wv, Wv_t);
  cvt_w_transpose<<<(DM * DM) / 256, 256, 0, stream>>>(Wo, Wo_t);

  const int projBlocks = (int)((ML / 16) * (DM / 16) / 8); // 4096
  proj_gemm<<<projBlocks, 256, 0, stream>>>(xq_bf,  Wq_t, bq, Qh, nullptr, (int)ML, 1);
  proj_gemm<<<projBlocks, 256, 0, stream>>>(xkv_bf, Wk_t, bk, Kh, nullptr, (int)ML, 1);
  proj_gemm<<<projBlocks, 256, 0, stream>>>(xkv_bf, Wv_t, bv, Vh, nullptr, (int)ML, 1);

  const int nCorr = BB * NH * LSEQ;            // 131072
  zero_f32<<<nCorr / 256, 256, 0, stream>>>(corr, nCorr);
  corr_kernel<<<BB * NH * 32, 256, 0, stream>>>(Qh, Kh, corr);

  topk_softmax<<<BB * NH, 256, 0, stream>>>(corr, attn, delays);
  gather_ctx<<<nX / 256, 256, 0, stream>>>(Vh, attn, delays, ctx);

  proj_gemm<<<projBlocks, 256, 0, stream>>>(ctx, Wo_t, bo, nullptr, out, (int)ML, 0);
}
